// CrossAttention_25701084299424
// MI455X (gfx1250) — matrix operand order repaired and checked by reading, not compiled, not run on hardware
//
#include <hip/hip_runtime.h>

typedef __attribute__((ext_vector_type(16))) _Float16 v16h;
typedef __attribute__((ext_vector_type(8)))  float    v8f;
typedef _Float16 half_t;

static constexpr int   kB     = 8;
static constexpr int   kHW    = 4096;   // 64*64 tokens per batch
static constexpr int   kCin   = 128;
static constexpr int   kDim   = 64;
static constexpr float kScale = 0.125f; // 1/sqrt(64), folded into Q
static constexpr float kLog2e = 1.44269504088896340736f;
static constexpr float kEps   = 1e-5f;

// LDS row strides padded to avoid 64-bank conflicts (see analysis):
static constexpr int kTS = 72;   // 16-bit tiles: dword-stride 36 -> disjoint banks
static constexpr int kWS = 136;  // qkv weight rows: dword-stride 68 -> bank 4m
static constexpr int kYS = 68;   // f32 staging rows: bank 4m per lane

static __device__ inline v8f wmma16(v16h a, v16h b, v8f c) {
  return __builtin_amdgcn_wmma_f32_16x16x32_f16(false, a, false, b,
                                                (short)0, c, false, false);
}
static __device__ inline v8f vzero8() {
  v8f z;
  #pragma unroll
  for (int r = 0; r < 8; ++r) z[r] = 0.f;
  return z;
}
// B fragment (32x16, col = lane&15, K = (lane>>4)*16 + e): 32 contiguous bytes.
static __device__ inline v16h ldB(const half_t* p) {
  v16h b;
  ((uint4*)&b)[0] = *(const uint4*)(p);
  ((uint4*)&b)[1] = *(const uint4*)(p + 8);
  return b;
}
// A fragment (16x32, row = lane&15, K = (lane>>4)*16 + e): 32 contiguous bytes.
static __device__ inline v16h ldA(const half_t* row, int kc, int hi) {
  v16h a;
  ((uint4*)&a)[0] = *(const uint4*)(row + kc * 32 + hi * 16);
  ((uint4*)&a)[1] = *(const uint4*)(row + kc * 32 + hi * 16 + 8);
  return a;
}

// ---------------------------------------------------------------------------
// K0: weights f32 -> f16 (row-major (o,c) is already B-fragment friendly)
// ---------------------------------------------------------------------------
__global__ __launch_bounds__(256) void cvt_weights_kernel(
    const float* __restrict__ Wq, const float* __restrict__ Wkv,
    const float* __restrict__ Wout,
    half_t* __restrict__ Wqh, half_t* __restrict__ Wkvh, half_t* __restrict__ Wouth)
{
  const int gid = blockIdx.x * 256 + threadIdx.x;
  const int nq = kDim * kCin, nkv = 2 * kDim * kCin, no = kCin * kDim;
  if (gid < nq)                 Wqh[gid]              = (half_t)Wq[gid];
  else if (gid < nq + nkv)      Wkvh[gid - nq]        = (half_t)Wkv[gid - nq];
  else if (gid < nq + nkv + no) Wouth[gid - nq - nkv] = (half_t)Wout[gid - nq - nkv];
}

// ---------------------------------------------------------------------------
// K1: channel LayerNorm for x and context -> f16 token-major (tok, 128)
// ---------------------------------------------------------------------------
__global__ __launch_bounds__(256) void ln_kernel(
    const float* __restrict__ x,    const float* __restrict__ ctx,
    const float* __restrict__ lnxw, const float* __restrict__ lnxb,
    const float* __restrict__ lncw, const float* __restrict__ lncb,
    half_t* __restrict__ Xn, half_t* __restrict__ Cn)
{
  const int gid = blockIdx.x * 256 + threadIdx.x;
  const int b   = gid / kHW;
  const int pos = gid - b * kHW;

  float xr[kCin];
  #pragma unroll 2
  for (int pass = 0; pass < 2; ++pass) {
    const float* src = (pass == 0 ? x : ctx) + (size_t)b * kCin * kHW + pos;
    const float* gw  = pass == 0 ? lnxw : lncw;
    const float* gb  = pass == 0 ? lnxb : lncb;
    half_t*      dst = (pass == 0 ? Xn : Cn) + (size_t)gid * kCin;

    float mu = 0.f;
    #pragma unroll
    for (int c = 0; c < kCin; ++c) { xr[c] = src[(size_t)c * kHW]; mu += xr[c]; }
    mu *= (1.f / kCin);
    float var = 0.f;
    #pragma unroll
    for (int c = 0; c < kCin; ++c) { float d = xr[c] - mu; var += d * d; }
    var *= (1.f / kCin);
    const float rs = rsqrtf(var + kEps);
    #pragma unroll
    for (int c = 0; c < kCin; ++c)
      dst[c] = (half_t)((xr[c] - mu) * rs * gw[c] + gb[c]);
  }
}

// ---------------------------------------------------------------------------
// K2: WMMA GEMM:  Q = relu(Xn Wq^T + bq)*scale,  [K|V] = relu(Cn Wkv^T + bkv)
// Weights staged once in LDS (padded rows); kc-outer / nb-inner gives four
// independent WMMA chains with ds_load latency overlapped.
//   Q,K: (b, token, 64) f16    V: (b, d, token) f16 (transposed)
// ---------------------------------------------------------------------------
__global__ __launch_bounds__(128) void qkv_gemm_kernel(
    const half_t* __restrict__ Xn,  const half_t* __restrict__ Cn,
    const half_t* __restrict__ Wqh, const half_t* __restrict__ Wkvh,
    const float* __restrict__ bq,   const float* __restrict__ bkv,
    half_t* __restrict__ Qh, half_t* __restrict__ Kh, half_t* __restrict__ Vh)
{
  __shared__ half_t Wls[192 * kWS];  // rows 0..63 Wq, 64..191 Wkv  (~52 KB)
  __shared__ half_t Qs[64 * 64];     // [tok][d]
  __shared__ half_t Ks[64 * 64];     // [tok][d]
  __shared__ half_t Vs[64 * kTS];    // [d][tok], padded

  const int t0   = blockIdx.x * 64;
  const int b    = t0 >> 12;
  const int ti   = t0 & (kHW - 1);
  const int wave = threadIdx.x >> 5, lane = threadIdx.x & 31;
  const int m    = lane & 15, hi = lane >> 4;
  const int arow = wave * 16 + m;

  // stage weights in LDS with padded rows
  for (int i = threadIdx.x; i < 64 * 16; i += 128) {
    const int row = i >> 4, seg = i & 15;
    *(uint4*)(&Wls[row * kWS + seg * 8]) = ((const uint4*)Wqh)[i];
  }
  for (int i = threadIdx.x; i < 128 * 16; i += 128) {
    const int row = i >> 4, seg = i & 15;
    *(uint4*)(&Wls[(64 + row) * kWS + seg * 8]) = ((const uint4*)Wkvh)[i];
  }

  // A fragments for both streams: 16 rows x 128, 4 K-chunks each
  v16h ax[4], ac[4];
  {
    const half_t* xp = Xn + (size_t)(t0 + arow) * kCin;
    const half_t* cp = Cn + (size_t)(t0 + arow) * kCin;
    #pragma unroll
    for (int kc = 0; kc < 4; ++kc) { ax[kc] = ldA(xp, kc, hi); ac[kc] = ldA(cp, kc, hi); }
  }
  __syncthreads();

  // ---- Q ----
  {
    v8f c[4];
    #pragma unroll
    for (int nb = 0; nb < 4; ++nb) c[nb] = vzero8();
    #pragma unroll
    for (int kc = 0; kc < 4; ++kc)
      #pragma unroll
      for (int nb = 0; nb < 4; ++nb)
        c[nb] = wmma16(ax[kc], ldB(&Wls[(nb * 16 + m) * kWS + kc * 32 + hi * 16]), c[nb]);
    #pragma unroll
    for (int nb = 0; nb < 4; ++nb) {
      const float bias = bq[nb * 16 + m];
      #pragma unroll
      for (int r = 0; r < 8; ++r)
        Qs[(wave * 16 + r + hi * 8) * 64 + nb * 16 + m] =
            (half_t)(fmaxf(c[nb][r] + bias, 0.f) * kScale);
    }
  }
  // ---- K ----
  {
    v8f c[4];
    #pragma unroll
    for (int nb = 0; nb < 4; ++nb) c[nb] = vzero8();
    #pragma unroll
    for (int kc = 0; kc < 4; ++kc)
      #pragma unroll
      for (int nb = 0; nb < 4; ++nb)
        c[nb] = wmma16(ac[kc], ldB(&Wls[(64 + nb * 16 + m) * kWS + kc * 32 + hi * 16]), c[nb]);
    #pragma unroll
    for (int nb = 0; nb < 4; ++nb) {
      const float bias = bkv[nb * 16 + m];
      #pragma unroll
      for (int r = 0; r < 8; ++r)
        Ks[(wave * 16 + r + hi * 8) * 64 + nb * 16 + m] =
            (half_t)fmaxf(c[nb][r] + bias, 0.f);
    }
  }
  // ---- V (transposed into Vs[d][tok]) ----
  {
    v8f c[4];
    #pragma unroll
    for (int nb = 0; nb < 4; ++nb) c[nb] = vzero8();
    #pragma unroll
    for (int kc = 0; kc < 4; ++kc)
      #pragma unroll
      for (int nb = 0; nb < 4; ++nb)
        c[nb] = wmma16(ac[kc], ldB(&Wls[(128 + nb * 16 + m) * kWS + kc * 32 + hi * 16]), c[nb]);
    #pragma unroll
    for (int nb = 0; nb < 4; ++nb) {
      const float bias = bkv[kDim + nb * 16 + m];
      #pragma unroll
      for (int r = 0; r < 8; ++r)
        Vs[(nb * 16 + m) * kTS + wave * 16 + r + hi * 8] =
            (half_t)fmaxf(c[nb][r] + bias, 0.f);
    }
  }

  __syncthreads();
  // coalesced stores
  half_t* qg = Qh + (size_t)t0 * kDim;
  half_t* kg = Kh + (size_t)t0 * kDim;
  for (int i = threadIdx.x; i < 64 * 8; i += 128) {
    *(uint4*)(qg + i * 8) = ((const uint4*)Qs)[i];
    *(uint4*)(kg + i * 8) = ((const uint4*)Ks)[i];
  }
  for (int i = threadIdx.x; i < 64 * 8; i += 128) {
    const int d = i >> 3, seg = i & 7;
    *(uint4*)(Vh + ((size_t)b * kDim + d) * kHW + ti + seg * 8) =
        *(const uint4*)(&Vs[d * kTS + seg * 8]);
  }
}

// ---------------------------------------------------------------------------
// K3: flash attention, double-buffered KV tiles (padded rows), WMMA f16.
// ---------------------------------------------------------------------------
__global__ __launch_bounds__(128) void flash_attn_kernel(
    const half_t* __restrict__ Qh, const half_t* __restrict__ Kh,
    const half_t* __restrict__ Vh, half_t* __restrict__ Oh)
{
  __shared__ half_t Kbuf[2][64 * kTS];   // [key tok][d]
  __shared__ half_t Vbuf[2][64 * kTS];   // [d][key tok]
  __shared__ half_t Pb[4][16 * kTS];     // per-wave P scratch

  const int b    = blockIdx.x >> 6;
  const int qt   = blockIdx.x & 63;
  const int wave = threadIdx.x >> 5, lane = threadIdx.x & 31;
  const int m    = lane & 15, hi = lane >> 4;

  // Q tile -> two 16x32 A fragments, held in registers
  v16h qa[2];
  {
    const half_t* qp = Qh + ((size_t)b * kHW + qt * 64 + wave * 16 + m) * kDim;
    #pragma unroll
    for (int kc = 0; kc < 2; ++kc) qa[kc] = ldA(qp, kc, hi);
  }

  v8f acc[4];
  float mrow[8], lpart[8];          // lpart = per-lane partial row sum
  #pragma unroll
  for (int nb = 0; nb < 4; ++nb) acc[nb] = vzero8();
  #pragma unroll
  for (int r = 0; r < 8; ++r) { mrow[r] = -3.0e38f; lpart[r] = 0.f; }

  // prologue: tile 0 -> buffer 0
  {
    const half_t* kg = Kh + (size_t)b * kHW * kDim;
    const half_t* vg = Vh + (size_t)b * kDim * kHW;
    #pragma unroll
    for (int j = 0; j < 4; ++j) {
      const int i = threadIdx.x + j * 128, row = i >> 3, seg = i & 7;
      *(uint4*)(&Kbuf[0][row * kTS + seg * 8]) = *(const uint4*)(kg + row * 64 + seg * 8);
      *(uint4*)(&Vbuf[0][row * kTS + seg * 8]) = *(const uint4*)(vg + (size_t)row * kHW + seg * 8);
    }
  }
  __syncthreads();

  for (int kt = 0; kt < kHW / 64; ++kt) {
    const int cur = kt & 1, nxt = cur ^ 1;
    const bool more = (kt + 1 < kHW / 64);

    // issue next tile's global loads early (hidden behind WMMA + softmax)
    uint4 nk[4], nv[4];
    if (more) {
      const half_t* kg = Kh + ((size_t)b * kHW + (kt + 1) * 64) * kDim;
      const half_t* vg = Vh + (size_t)b * kDim * kHW + (kt + 1) * 64;
      #pragma unroll
      for (int j = 0; j < 4; ++j) {
        const int i = threadIdx.x + j * 128, row = i >> 3, seg = i & 7;
        nk[j] = *(const uint4*)(kg + row * 64 + seg * 8);
        nv[j] = *(const uint4*)(vg + (size_t)row * kHW + seg * 8);
      }
    }

    // ---- S = Q K^T : four 16x16 f32 accumulators ----
    v8f s[4];
    #pragma unroll
    for (int nb = 0; nb < 4; ++nb) {
      v8f c = vzero8();
      #pragma unroll
      for (int kc = 0; kc < 2; ++kc)
        c = wmma16(qa[kc], ldB(&Kbuf[cur][(nb * 16 + m) * kTS + kc * 32 + hi * 16]), c);
      s[nb] = c;
    }

    // ---- online softmax: row max reduce, rescale, exp ----
    #pragma unroll
    for (int r = 0; r < 8; ++r) {
      float v = s[0][r];
      #pragma unroll
      for (int nb = 1; nb < 4; ++nb) v = fmaxf(v, s[nb][r]);
      #pragma unroll
      for (int off = 1; off < 16; off <<= 1) v = fmaxf(v, __shfl_xor(v, off, 16));
      const float mnew = fmaxf(mrow[r], v);
      const float corr = exp2f((mrow[r] - mnew) * kLog2e);
      mrow[r] = mnew;
      lpart[r] *= corr;
      #pragma unroll
      for (int nb = 0; nb < 4; ++nb) acc[nb][r] *= corr;
    }
    #pragma unroll
    for (int nb = 0; nb < 4; ++nb)
      #pragma unroll
      for (int r = 0; r < 8; ++r) {
        const float p = exp2f((s[nb][r] - mrow[r]) * kLog2e);
        lpart[r] += p;
        Pb[wave][(r + hi * 8) * kTS + nb * 16 + m] = (half_t)p;
      }

    // ---- O += P V ----
    v16h pa[2];
    {
      const half_t* pp = &Pb[wave][m * kTS];
      #pragma unroll
      for (int kc = 0; kc < 2; ++kc) pa[kc] = ldA(pp, kc, hi);
    }
    #pragma unroll
    for (int nb = 0; nb < 4; ++nb) {
      v8f c = acc[nb];
      #pragma unroll
      for (int kc = 0; kc < 2; ++kc)
        c = wmma16(pa[kc], ldB(&Vbuf[cur][(nb * 16 + m) * kTS + kc * 32 + hi * 16]), c);
      acc[nb] = c;
    }

    // drain next tile into the alternate buffer
    if (more) {
      #pragma unroll
      for (int j = 0; j < 4; ++j) {
        const int i = threadIdx.x + j * 128, row = i >> 3, seg = i & 7;
        *(uint4*)(&Kbuf[nxt][row * kTS + seg * 8]) = nk[j];
        *(uint4*)(&Vbuf[nxt][row * kTS + seg * 8]) = nv[j];
      }
    }
    __syncthreads();
  }

  // ---- epilogue: single cross-lane l reduction, f16 output (token, 64) ----
  half_t* og = Oh + ((size_t)b * kHW + qt * 64 + wave * 16) * kDim;
  #pragma unroll
  for (int r = 0; r < 8; ++r) {
    float v = lpart[r];
    #pragma unroll
    for (int off = 1; off < 16; off <<= 1) v += __shfl_xor(v, off, 16);
    const float inv = 1.f / v;
    #pragma unroll
    for (int nb = 0; nb < 4; ++nb)
      og[(r + hi * 8) * kDim + nb * 16 + m] = (half_t)(acc[nb][r] * inv);
  }
}

// ---------------------------------------------------------------------------
// K4: WMMA GEMM: Y[b,o,pos] = O[b,pos,:] . Wout[o,:] + bout[o]
// Wout staged in LDS (padded); 8 independent chains; coalesced f32 stores.
// ---------------------------------------------------------------------------
__global__ __launch_bounds__(128) void out_gemm_kernel(
    const half_t* __restrict__ Oh, const half_t* __restrict__ Wouth,
    const float* __restrict__ bout, float* __restrict__ Y)
{
  __shared__ half_t Wos[kCin * kTS];   // padded weight rows (~18 KB)
  __shared__ float  Ys[kCin * kYS];    // [o][tok] padded (~35 KB)

  const int t0   = blockIdx.x * 64;
  const int b    = t0 >> 12;
  const int ti   = t0 & (kHW - 1);
  const int wave = threadIdx.x >> 5, lane = threadIdx.x & 31;
  const int m    = lane & 15, hi = lane >> 4;
  const int arow = wave * 16 + m;

  for (int i = threadIdx.x; i < kCin * 8; i += 128) {
    const int row = i >> 3, seg = i & 7;
    *(uint4*)(&Wos[row * kTS + seg * 8]) = ((const uint4*)Wouth)[i];
  }

  v16h ao[2];
  {
    const half_t* op = Oh + (size_t)(t0 + arow) * kDim;
    #pragma unroll
    for (int kc = 0; kc < 2; ++kc) ao[kc] = ldA(op, kc, hi);
  }
  __syncthreads();

  v8f c[8];
  #pragma unroll
  for (int nb = 0; nb < 8; ++nb) c[nb] = vzero8();
  #pragma unroll
  for (int kc = 0; kc < 2; ++kc)
    #pragma unroll
    for (int nb = 0; nb < 8; ++nb)
      c[nb] = wmma16(ao[kc], ldB(&Wos[(nb * 16 + m) * kTS + kc * 32 + hi * 16]), c[nb]);

  #pragma unroll
  for (int nb = 0; nb < 8; ++nb) {
    const float bias = bout[nb * 16 + m];
    #pragma unroll
    for (int r = 0; r < 8; ++r)
      Ys[(nb * 16 + m) * kYS + wave * 16 + r + hi * 8] = c[nb][r] + bias;
  }

  __syncthreads();
  for (int i = threadIdx.x; i < kCin * 16; i += 128) {   // uint4 = 4 floats
    const int o = i >> 4, seg = i & 15;
    *(uint4*)(Y + ((size_t)b * kCin + o) * kHW + ti + seg * 4) =
        *(const uint4*)(&Ys[o * kYS + seg * 4]);
  }
}

// ---------------------------------------------------------------------------
extern "C" void kernel_launch(void* const* d_in, const int* in_sizes, int n_in,
                              void* d_out, int out_size, void* d_ws, size_t ws_size,
                              hipStream_t stream) {
  const float* x    = (const float*)d_in[0];
  const float* ctx  = (const float*)d_in[1];
  const float* lnxw = (const float*)d_in[2];
  const float* lnxb = (const float*)d_in[3];
  const float* lncw = (const float*)d_in[4];
  const float* lncb = (const float*)d_in[5];
  const float* Wq   = (const float*)d_in[6];
  const float* bq   = (const float*)d_in[7];
  const float* Wkv  = (const float*)d_in[8];
  const float* bkv  = (const float*)d_in[9];
  const float* Wout = (const float*)d_in[10];
  const float* bout = (const float*)d_in[11];
  float* Y = (float*)d_out;

  char* ws = (char*)d_ws;
  const size_t tok = (size_t)kB * kHW;                 // 32768 tokens
  size_t off = 0;
  half_t* Wqh   = (half_t*)(ws + off); off += (size_t)kDim * kCin * 2;      // 16 KB
  half_t* Wkvh  = (half_t*)(ws + off); off += (size_t)2 * kDim * kCin * 2;  // 32 KB
  half_t* Wouth = (half_t*)(ws + off); off += (size_t)kCin * kDim * 2;      // 16 KB
  off = (off + 255) & ~(size_t)255;
  half_t* Xn = (half_t*)(ws + off); off += tok * kCin * 2;   // 8 MB
  half_t* Cn = (half_t*)(ws + off); off += tok * kCin * 2;   // 8 MB
  half_t* Qh = (half_t*)(ws + off); off += tok * kDim * 2;   // 4 MB
  half_t* Kh = (half_t*)(ws + off); off += tok * kDim * 2;   // 4 MB
  half_t* Vh = (half_t*)(ws + off); off += tok * kDim * 2;   // 4 MB
  half_t* Oh = (half_t*)(ws + off); off += tok * kDim * 2;   // 4 MB

  cvt_weights_kernel<<<(kDim * kCin + 2 * kDim * kCin + kCin * kDim + 255) / 256,
                       256, 0, stream>>>(Wq, Wkv, Wout, Wqh, Wkvh, Wouth);
  ln_kernel<<<(int)(tok / 256), 256, 0, stream>>>(x, ctx, lnxw, lnxb, lncw, lncb, Xn, Cn);
  qkv_gemm_kernel<<<(int)(tok / 64), 128, 0, stream>>>(Xn, Cn, Wqh, Wkvh, bq, bkv,
                                                       Qh, Kh, Vh);
  flash_attn_kernel<<<kB * (kHW / 64), 128, 0, stream>>>(Qh, Kh, Vh, Oh);
  out_gemm_kernel<<<(int)(tok / 64), 128, 0, stream>>>(Oh, Wouth, bout, Y);
}
